// CausalPredictor_14216341749932
// MI455X (gfx1250) — matrix operand order repaired and checked by reading, not compiled, not run on hardware
//
#include <hip/hip_runtime.h>
#include <hip/hip_bf16.h>

typedef __attribute__((ext_vector_type(16))) _Float16 v16h;
typedef __attribute__((ext_vector_type(8)))  _Float16 v8h;
typedef __attribute__((ext_vector_type(8)))  float    v8f;

namespace {
constexpr int kB      = 4;
constexpr int kSeq    = 1024;
constexpr int kDim    = 768;
constexpr int kHeads  = 12;
constexpr int kDHead  = 64;
constexpr int kMlp    = 3072;
constexpr int kDepth  = 6;
constexpr int kInner  = 768;
constexpr int kNFrame = 16;
constexpr int kRows   = kB * kSeq;     // 4096
constexpr int kQkvN   = 3 * kInner;    // 2304
constexpr float kScale = 0.125f;       // 64^-0.5
}

// --- CDNA5 async global->LDS path (guarded; falls back to register relay) ---
#if defined(__has_builtin)
#if __has_builtin(__builtin_amdgcn_global_load_async_to_lds_b128) && \
    __has_builtin(__builtin_amdgcn_s_wait_asynccnt)
#define USE_ASYNC_LDS 1
#endif
#endif
#ifndef USE_ASYNC_LDS
#define USE_ASYNC_LDS 0
#endif

#if USE_ASYNC_LDS
typedef int v4i_ __attribute__((vector_size(16)));
typedef __attribute__((address_space(1))) v4i_ gv4i;  // global int4
typedef __attribute__((address_space(3))) v4i_ lv4i;  // LDS int4
#endif

__device__ __forceinline__ void stage16(const _Float16* gsrc, _Float16* ldst) {
#if USE_ASYNC_LDS
  __builtin_amdgcn_global_load_async_to_lds_b128((gv4i*)gsrc, (lv4i*)ldst, 0, 0);
#else
  *(v8h*)ldst = *(const v8h*)gsrc;
#endif
}

__device__ __forceinline__ void wait_stage() {
#if USE_ASYNC_LDS
  __builtin_amdgcn_s_wait_asynccnt(0);
#endif
}

__device__ __forceinline__ v8f zero8() {
  v8f z = {0.f, 0.f, 0.f, 0.f, 0.f, 0.f, 0.f, 0.f};
  return z;
}

// Build a 16x32 f16 A-fragment. p points at (row, k0 + kb) where
// row = m0 + (lane&15), kb = (lane>>4)*16. Lane elements [0..15] = K kb..kb+15
// (contiguous order, matching the packed B fragments).
__device__ __forceinline__ v16h make_a(const _Float16* p) {
  v8h lo = *(const v8h*)p;
  v8h hi = *(const v8h*)(p + 8);
  v16h a;
#pragma unroll
  for (int i = 0; i < 8; ++i) { a[i] = lo[i]; a[8 + i] = hi[i]; }
  return a;
}

__device__ __forceinline__ v8f wmma_f16(v16h a, v16h b, v8f c) {
  return __builtin_amdgcn_wmma_f32_16x16x32_f16(false, a, false, b, (short)0, c,
                                                false, false);
}

// ---------------------------------------------------------------------------
// x = x_in + pos (broadcast over batch)
// ---------------------------------------------------------------------------
__global__ void add_pos_kernel(const float* __restrict__ x,
                               const float* __restrict__ pos,
                               float* __restrict__ xo) {
  int i = blockIdx.x * 256 + threadIdx.x;
  if (i < kRows * kDim) {
    int sd = i % (kSeq * kDim);
    xo[i] = x[i] + pos[sd];
  }
}

// ---------------------------------------------------------------------------
// Pack f32 weight [K,N] row-major into B-fragment-major f16:
// tile t = (k0/32)*(N/16) + (n0/16); within tile: lane*16 + j where
// lane = ((k>>4)&1)*16 + (n&15), j = k&15  (contiguous 32B per lane on load)
// ---------------------------------------------------------------------------
__global__ void pack_w_kernel(const float* __restrict__ W,
                              _Float16* __restrict__ Wp, int Ktot, int N) {
  int gid = blockIdx.x * 256 + threadIdx.x;
  int lane = gid & 31;
  int t = gid >> 5;
  int ntiles = N >> 4;
  int ktiles = Ktot >> 5;
  if (t >= ktiles * ntiles) return;
  int k0 = (t / ntiles) * 32;
  int n0 = (t % ntiles) * 16;
  int n = n0 + (lane & 15);
  int kb = k0 + (lane >> 4) * 16;
  _Float16* dst = Wp + (size_t)t * 512 + lane * 16;
  const float* src = W + (size_t)kb * N + n;
#pragma unroll
  for (int j = 0; j < 16; ++j) dst[j] = (_Float16)src[(size_t)j * N];
}

// ---------------------------------------------------------------------------
// Pack V third of qkv into B-fragment-major layout per (b,h): V is [S, 64]
// ---------------------------------------------------------------------------
__global__ void pack_v_kernel(const _Float16* __restrict__ qkv,
                              _Float16* __restrict__ vpack) {
  int gid = blockIdx.x * 256 + threadIdx.x;
  int lane = gid & 31;
  int t = gid >> 5;
  if (t >= kB * kHeads * (kSeq / 32) * 4) return;
  int nf = t & 3;              // d tile (16 wide)
  int kt = (t >> 2) & 31;      // key tile (32 keys)
  int bh = t >> 7;             // b*kHeads + h
  int h = bh % kHeads;
  int b = bh / kHeads;
  int d = nf * 16 + (lane & 15);
  int key = kt * 32 + (lane >> 4) * 16;
  const _Float16* src =
      qkv + (size_t)(b * kSeq + key) * kQkvN + 2 * kInner + h * kDHead + d;
  _Float16* dst = vpack + (size_t)bh * kSeq * kDHead +
                  (size_t)(kt * 4 + nf) * 512 + lane * 16;
#pragma unroll
  for (int j = 0; j < 16; ++j) dst[j] = src[(size_t)j * kQkvN];
}

// ---------------------------------------------------------------------------
// LayerNorm over rows of 768; writes f16 (outh) or f32 (outf)
// ---------------------------------------------------------------------------
__global__ __launch_bounds__(256, 1) void layernorm_kernel(
    const float* __restrict__ x, const float* __restrict__ w,
    const float* __restrict__ bwt, _Float16* outh, float* outf) {
  const int row = blockIdx.x;
  const int tid = threadIdx.x;
  const float* xr = x + (size_t)row * kDim;
  float v0 = xr[tid], v1 = xr[tid + 256], v2 = xr[tid + 512];
  __shared__ float rsum[256], rsq[256];
  rsum[tid] = v0 + v1 + v2;
  rsq[tid] = v0 * v0 + v1 * v1 + v2 * v2;
  __syncthreads();
  for (int off = 128; off > 0; off >>= 1) {
    if (tid < off) { rsum[tid] += rsum[tid + off]; rsq[tid] += rsq[tid + off]; }
    __syncthreads();
  }
  const float mu = rsum[0] * (1.0f / kDim);
  const float var = rsq[0] * (1.0f / kDim) - mu * mu;
  const float rstd = rsqrtf(var + 1e-5f);
  float vv[3] = {v0, v1, v2};
#pragma unroll
  for (int i = 0; i < 3; ++i) {
    int c = tid + i * 256;
    float y = (vv[i] - mu) * rstd * w[c] + bwt[c];
    if (outh) outh[(size_t)row * kDim + c] = (_Float16)y;
    else      outf[(size_t)row * kDim + c] = y;
  }
}

// ---------------------------------------------------------------------------
// WMMA GEMM with double-buffered async B staging through LDS.
// C[M,N] = A[M,K](f16,row-major,lda) x Bpack(K,N fragment-packed).
// 8 waves/block; wave => 16(M) x 64(N). Per k-step the block shares one 4KB
// B tile: staged global->LDS with GLOBAL_LOAD_ASYNC_TO_LDS_B128 (ASYNCcnt),
// issued one iteration ahead. A-fragments are register-prefetched.
// Epilogue modes:
//  0: outh = (f16)acc                        (QKV, no bias)
//  1: outf = resid + acc + bias  (f32)       (projections w/ residual)
//  2: outh = (f16)gelu(acc + bias)           (MLP up)
// ---------------------------------------------------------------------------
__global__ __launch_bounds__(256, 1) void gemm_wmma_kernel(
    const _Float16* __restrict__ A, int lda, const _Float16* __restrict__ Bp,
    const float* __restrict__ bias, const float* resid, float* outf,
    _Float16* outh, int ldo, int N, int K, int mode) {
  const int tid = threadIdx.x;
  const int lane = tid & 31;
  const int wv = tid >> 5;
  const int m0 = blockIdx.y * 128 + wv * 16;
  const int n0 = blockIdx.x * 64;
  const int ntiles = N >> 4;

  __shared__ _Float16 bstage[2][2048] __attribute__((aligned(16)));  // 2 x 4KB

  v8f acc[4];
#pragma unroll
  for (int i = 0; i < 4; ++i) acc[i] = zero8();

  const int arow = m0 + (lane & 15);
  const _Float16* ap = A + (size_t)arow * lda + (lane >> 4) * 16;

  const size_t tileStride = (size_t)ntiles * 512;          // halves per k-step
  const _Float16* btile = Bp + (size_t)(n0 >> 4) * 512;    // this block's cols
  const int ksteps = K >> 5;

  // prologue: stage k-step 0, prefetch A fragment 0
  stage16(btile + (size_t)tid * 8, &bstage[0][tid * 8]);
  v16h a = make_a(ap);

  for (int ks = 0; ks < ksteps; ++ks) {
    const int cur = ks & 1;
    wait_stage();
    __syncthreads();  // staged tile visible to all waves

    v16h b0 = *(const v16h*)&bstage[cur][0 * 512 + lane * 16];
    v16h b1 = *(const v16h*)&bstage[cur][1 * 512 + lane * 16];
    v16h b2 = *(const v16h*)&bstage[cur][2 * 512 + lane * 16];
    v16h b3 = *(const v16h*)&bstage[cur][3 * 512 + lane * 16];
    __syncthreads();  // all waves done reading before next overwrite cycle

    if (ks + 1 < ksteps)
      stage16(btile + (size_t)(ks + 1) * tileStride + (size_t)tid * 8,
              &bstage[cur ^ 1][tid * 8]);
    v16h a_next = (ks + 1 < ksteps) ? make_a(ap + (size_t)(ks + 1) * 32) : a;

    acc[0] = wmma_f16(a, b0, acc[0]);
    acc[1] = wmma_f16(a, b1, acc[1]);
    acc[2] = wmma_f16(a, b2, acc[2]);
    acc[3] = wmma_f16(a, b3, acc[3]);
    a = a_next;
  }

  const int nc = lane & 15;
  const int rbase = (lane >> 4) * 8;
#pragma unroll
  for (int nf = 0; nf < 4; ++nf) {
#pragma unroll
    for (int r = 0; r < 8; ++r) {
      int row = m0 + rbase + r;
      int col = n0 + nf * 16 + nc;
      float v = acc[nf][r];
      if (mode == 0) {
        outh[(size_t)row * ldo + col] = (_Float16)v;
      } else if (mode == 1) {
        outf[(size_t)row * ldo + col] =
            resid[(size_t)row * ldo + col] + v + bias[col];
      } else {
        float g = v + bias[col];
        g = 0.5f * g * (1.0f + erff(g * 0.70710678118654752f));
        outh[(size_t)row * ldo + col] = (_Float16)g;
      }
    }
  }
}

// ---------------------------------------------------------------------------
// Block-causal flash attention. Grid: (frame, head, batch); 128 thr = 4 waves.
// Each wave: 16 q rows x 64 d out, streaming over key tiles of 32 with
// online softmax. Frame f attends keys [0, (f+1)*64) -- no intra-frame mask.
// ---------------------------------------------------------------------------
__global__ __launch_bounds__(128, 1) void attn_wmma_kernel(
    const _Float16* __restrict__ qkv, const _Float16* __restrict__ vpack,
    _Float16* __restrict__ attn) {
  const int f = blockIdx.x;
  const int h = blockIdx.y;
  const int b = blockIdx.z;
  const int lane = threadIdx.x & 31;
  const int wv = threadIdx.x >> 5;
  __shared__ _Float16 plds[4][16][32] __attribute__((aligned(16)));

  const int m0 = f * 64 + wv * 16;
  const _Float16* qbase = qkv + (size_t)(b * kSeq) * kQkvN + h * kDHead;
  const _Float16* kbase = qbase + kInner;
  const _Float16* vbase = vpack + (size_t)(b * kHeads + h) * kSeq * kDHead;

  v16h qa0, qa1;  // Q A-fragments for d=0..31 and d=32..63
  {
    int row = m0 + (lane & 15);
    const _Float16* p = qbase + (size_t)row * kQkvN + (lane >> 4) * 16;
    qa0 = make_a(p);
    qa1 = make_a(p + 32);
  }

  float mrow[8], lrow[8];
  v8f acc[4];
#pragma unroll
  for (int r = 0; r < 8; ++r) { mrow[r] = -3.0e38f; lrow[r] = 0.f; }
#pragma unroll
  for (int i = 0; i < 4; ++i) acc[i] = zero8();

  const int nkt = (f + 1) * 2;  // key tiles of 32
  for (int kt = 0; kt < nkt; ++kt) {
    const int key0 = kt * 32;
    v8f s0 = zero8(), s1 = zero8();
    {
      // B-fragment of K^T: element (k=d, n=key) = K[key][d], contiguous in d
      const _Float16* kp =
          kbase + (size_t)(key0 + (lane & 15)) * kQkvN + (lane >> 4) * 16;
      v16h klo0 = *(const v16h*)kp;                        // keys 0-15, d 0-31
      v16h klo1 = *(const v16h*)(kp + 32);                 // keys 0-15, d 32-63
      v16h khi0 = *(const v16h*)(kp + (size_t)16 * kQkvN); // keys 16-31
      v16h khi1 = *(const v16h*)(kp + (size_t)16 * kQkvN + 32);
      s0 = wmma_f16(qa0, klo0, s0);
      s0 = wmma_f16(qa1, klo1, s0);
      s1 = wmma_f16(qa0, khi0, s1);
      s1 = wmma_f16(qa1, khi1, s1);
    }
    // online softmax: rows split as lanes 0-15 -> M=r, lanes 16-31 -> M=8+r
    float p0[8], p1[8];
#pragma unroll
    for (int r = 0; r < 8; ++r) {
      float a0 = s0[r] * kScale;
      float a1 = s1[r] * kScale;
      float mx = fmaxf(a0, a1);
      mx = fmaxf(mx, __shfl_xor(mx, 1));
      mx = fmaxf(mx, __shfl_xor(mx, 2));
      mx = fmaxf(mx, __shfl_xor(mx, 4));
      mx = fmaxf(mx, __shfl_xor(mx, 8));
      float mn = fmaxf(mrow[r], mx);
      float corr = __expf(mrow[r] - mn);
      mrow[r] = mn;
      p0[r] = __expf(a0 - mn);
      p1[r] = __expf(a1 - mn);
      float rs = p0[r] + p1[r];
      rs += __shfl_xor(rs, 1);
      rs += __shfl_xor(rs, 2);
      rs += __shfl_xor(rs, 4);
      rs += __shfl_xor(rs, 8);
      lrow[r] = lrow[r] * corr + rs;
#pragma unroll
      for (int nf = 0; nf < 4; ++nf) acc[nf][r] *= corr;
    }
    // re-layout P (C-fragment) -> A-fragment via wave-private LDS
    {
      int prow = (lane >> 4) * 8;
      int pc = lane & 15;
#pragma unroll
      for (int r = 0; r < 8; ++r) {
        plds[wv][prow + r][pc] = (_Float16)p0[r];
        plds[wv][prow + r][16 + pc] = (_Float16)p1[r];
      }
    }
    v16h pa = make_a(&plds[wv][lane & 15][(lane >> 4) * 16]);
    const _Float16* vt = vbase + (size_t)(kt * 4) * 512 + lane * 16;
#pragma unroll
    for (int nf = 0; nf < 4; ++nf) {
      v16h vb = *(const v16h*)(vt + (size_t)nf * 512);
      acc[nf] = wmma_f16(pa, vb, acc[nf]);
    }
  }

  _Float16* obase = attn + (size_t)(b * kSeq) * kInner + h * kDHead;
  const int nc = lane & 15;
  const int rbase = (lane >> 4) * 8;
#pragma unroll
  for (int r = 0; r < 8; ++r) {
    float inv = 1.0f / lrow[r];
    int row = m0 + rbase + r;
#pragma unroll
    for (int nf = 0; nf < 4; ++nf)
      obase[(size_t)row * kInner + nf * 16 + nc] = (_Float16)(acc[nf][r] * inv);
  }
}

// ---------------------------------------------------------------------------
extern "C" void kernel_launch(void* const* d_in, const int* in_sizes, int n_in,
                              void* d_out, int out_size, void* d_ws,
                              size_t ws_size, hipStream_t stream) {
  (void)in_sizes; (void)n_in; (void)out_size;
  const float* x_in = (const float*)d_in[0];
  const float* pos  = (const float*)d_in[1];
  const float* ln1w = (const float*)d_in[2];
  const float* ln1b = (const float*)d_in[3];
  const float* wqkv = (const float*)d_in[4];
  const float* wo   = (const float*)d_in[5];
  const float* bo   = (const float*)d_in[6];
  const float* ln2w = (const float*)d_in[7];
  const float* ln2b = (const float*)d_in[8];
  const float* w1   = (const float*)d_in[9];
  const float* b1   = (const float*)d_in[10];
  const float* w2   = (const float*)d_in[11];
  const float* b2   = (const float*)d_in[12];
  const float* lnfw = (const float*)d_in[13];
  const float* lnfb = (const float*)d_in[14];

  char* wsb = (char*)d_ws;
  size_t off = 0;
  auto take = [&](size_t bytes) -> void* {
    void* p = wsb + off;
    off = (off + bytes + 255) & ~(size_t)255;
    return p;
  };
  float*    xbuf  = (float*)take((size_t)kRows * kDim * 4);
  _Float16* hbuf  = (_Float16*)take((size_t)kRows * kDim * 2);
  _Float16* qkvb  = (_Float16*)take((size_t)kRows * kQkvN * 2);
  _Float16* vpk   = (_Float16*)take((size_t)kB * kHeads * kSeq * kDHead * 2);
  _Float16* attnb = (_Float16*)take((size_t)kRows * kInner * 2);
  _Float16* hmid  = (_Float16*)take((size_t)kRows * kMlp * 2);
  _Float16* wpk   = (_Float16*)take((size_t)kDim * kMlp * 2);
  if (off > ws_size) return;

  add_pos_kernel<<<(kRows * kDim + 255) / 256, 256, 0, stream>>>(x_in, pos, xbuf);

  for (int l = 0; l < kDepth; ++l) {
    // --- attention ---
    layernorm_kernel<<<kRows, 256, 0, stream>>>(xbuf, ln1w + l * kDim,
                                                ln1b + l * kDim, hbuf, nullptr);
    {
      int tiles = (kDim / 32) * (kQkvN / 16);
      pack_w_kernel<<<(tiles * 32 + 255) / 256, 256, 0, stream>>>(
          wqkv + (size_t)l * kDim * kQkvN, wpk, kDim, kQkvN);
      gemm_wmma_kernel<<<dim3(kQkvN / 64, kRows / 128), 256, 0, stream>>>(
          hbuf, kDim, wpk, nullptr, nullptr, nullptr, qkvb, kQkvN, kQkvN, kDim, 0);
    }
    {
      int total = kB * kHeads * (kSeq / 32) * 4 * 32;
      pack_v_kernel<<<(total + 255) / 256, 256, 0, stream>>>(qkvb, vpk);
    }
    attn_wmma_kernel<<<dim3(kNFrame, kHeads, kB), 128, 0, stream>>>(qkvb, vpk,
                                                                    attnb);
    {
      int tiles = (kInner / 32) * (kDim / 16);
      pack_w_kernel<<<(tiles * 32 + 255) / 256, 256, 0, stream>>>(
          wo + (size_t)l * kInner * kDim, wpk, kInner, kDim);
      gemm_wmma_kernel<<<dim3(kDim / 64, kRows / 128), 256, 0, stream>>>(
          attnb, kInner, wpk, bo + l * kDim, xbuf, xbuf, nullptr, kDim, kDim,
          kInner, 1);
    }
    // --- MLP ---
    layernorm_kernel<<<kRows, 256, 0, stream>>>(xbuf, ln2w + l * kDim,
                                                ln2b + l * kDim, hbuf, nullptr);
    {
      int tiles = (kDim / 32) * (kMlp / 16);
      pack_w_kernel<<<(tiles * 32 + 255) / 256, 256, 0, stream>>>(
          w1 + (size_t)l * kDim * kMlp, wpk, kDim, kMlp);
      gemm_wmma_kernel<<<dim3(kMlp / 64, kRows / 128), 256, 0, stream>>>(
          hbuf, kDim, wpk, b1 + l * kMlp, nullptr, nullptr, hmid, kMlp, kMlp,
          kDim, 2);
    }
    {
      int tiles = (kMlp / 32) * (kDim / 16);
      pack_w_kernel<<<(tiles * 32 + 255) / 256, 256, 0, stream>>>(
          w2 + (size_t)l * kMlp * kDim, wpk, kMlp, kDim);
      gemm_wmma_kernel<<<dim3(kDim / 64, kRows / 128), 256, 0, stream>>>(
          hmid, kMlp, wpk, b2 + l * kDim, xbuf, xbuf, nullptr, kDim, kDim, kMlp,
          1);
    }
  }
  layernorm_kernel<<<kRows, 256, 0, stream>>>(xbuf, lnfw, lnfb, nullptr,
                                              (float*)d_out);
}
